// QMamba_24524263260382
// MI455X (gfx1250) — matrix operand order repaired and checked by reading, not compiled, not run on hardware
//
#include <hip/hip_runtime.h>
#include <cstdint>
#include <cstddef>

// ---------------------------------------------------------------------------
// QMamba forward for MI455X (gfx1250, wave32, WMMA, async LDS staging).
//   0) one-time fp32->bf16 conversion of x, W_in, W_x, W_dt, W_out
//   1) xz    = x @ W_in^T                      (bf16 WMMA GEMM, fp32 accum)
//   2) u_b   = bf16(silu(conv(xz[:,:, :DI]) + conv_b))
//   3) x_dbl = u @ W_x^T   (fp32 out + bf16 mirror for dt_proj)
//   4) delta = softplus(x_dbl[:,:, :128] @ W_dt^T + b_dt)
//   5) chunked selective scan (reduce / prefix / apply) -> y_b (bf16, gated)
//   6) out   = y @ W_out^T
// GEMM staging uses GLOBAL_LOAD_ASYNC_TO_LDS_B128 (ASYNCcnt-tracked) when the
// builtin exists, else synchronous b128 copies. Tile-aligned call sites use
// the unguarded (BOUNDED=false) instantiation for a branch-free inner loop.
// ---------------------------------------------------------------------------

#define D_MODEL_C   2048
#define D_STATE_C   16
#define D_CONV_C    4
#define DT_RANK_C   128
#define D_INNER_C   4096
#define BATCH_C     2
#define SEQLEN_C    2048
#define CHUNKS_C    16
#define CLEN_C      (SEQLEN_C / CHUNKS_C)     // 128
#define NDBL_C      (DT_RANK_C + 2 * D_STATE_C)  // 160

typedef __attribute__((ext_vector_type(16))) __bf16 v16bf;
typedef __attribute__((ext_vector_type(8)))  float  v8f;

// Pointee type of the async-LDS builtins per hipcc diagnostic:
// "__attribute__((__vector_size__(4 * sizeof(int)))) int *"
typedef int v4i_vs __attribute__((vector_size(16)));
typedef __attribute__((address_space(1))) v4i_vs* gptr128;
typedef __attribute__((address_space(3))) v4i_vs* lptr128;

#if __has_builtin(__builtin_amdgcn_global_load_async_to_lds_b128)
#define ASYNC_COPY 1
#else
#define ASYNC_COPY 0
#endif

__device__ __forceinline__ void wait_async() {
#if __has_builtin(__builtin_amdgcn_s_wait_asynccnt)
  __builtin_amdgcn_s_wait_asynccnt(0);
#else
  asm volatile("s_wait_asynccnt 0x0" ::: "memory");
#endif
}

__device__ __forceinline__ unsigned short f2bf(float f) {
  unsigned int u = __builtin_bit_cast(unsigned int, f);
  u += 0x7FFFu + ((u >> 16) & 1u);          // round-to-nearest-even
  return (unsigned short)(u >> 16);
}
__device__ __forceinline__ float bf2f(unsigned short h) {
  unsigned int u = ((unsigned int)h) << 16;
  return __builtin_bit_cast(float, u);
}

struct Frag32 { uint4 lo, hi; };

// A-matrix 16x32 bf16 fragment (ISA 7.12.2): lane holds row m = lane%16;
// dwords 0..3: K = half*8 + [0..7]; dwords 4..7: K = 16 + half*8 + [0..7].
__device__ __forceinline__ v16bf load_fragA(const unsigned short* row, int half) {
  Frag32 f;
  f.lo = *(const uint4*)(row + half * 8);
  f.hi = *(const uint4*)(row + 16 + half * 8);
  return __builtin_bit_cast(v16bf, f);
}
// B-matrix 32x16 bf16 fragment: lane holds column n = lane%16; same K order
// as A: dwords 0..3: K = half*8 + [0..7]; dwords 4..7: K = 16 + half*8 + [0..7].
__device__ __forceinline__ v16bf load_fragB(const unsigned short* row, int half) {
  Frag32 f;
  f.lo = *(const uint4*)(row + half * 8);
  f.hi = *(const uint4*)(row + 16 + half * 8);
  return __builtin_bit_cast(v16bf, f);
}

// Copy one 16B chunk global(bf16) -> LDS. Async path uses ASYNCcnt-tracked
// GLOBAL_LOAD_ASYNC_TO_LDS_B128; generic->as3 via low-32-bit LDS offset
// (flat LDS aperture keeps the byte offset in addr[31:0], ISA 10.2).
__device__ __forceinline__ void stage_chunk(const unsigned short* __restrict__ gsrc,
                                            unsigned short* ldst, bool ok) {
#if ASYNC_COPY
  if (ok) {
    __builtin_amdgcn_global_load_async_to_lds_b128(
        (gptr128)(uintptr_t)gsrc,
        (lptr128)(unsigned int)(uintptr_t)ldst,
        0, 0);
  } else {
    *(uint4*)ldst = make_uint4(0u, 0u, 0u, 0u);
  }
#else
  uint4 v;
  if (ok) v = *(const uint4*)gsrc;
  else    v = make_uint4(0u, 0u, 0u, 0u);
  *(uint4*)ldst = v;
#endif
}

// C[M,N] = A[M,K] @ W[N,K]^T ; A,W bf16, C fp32 (WMMA bf16, fp32 accum).
// ACT==1: C = softplus(C + bias[col]).  MIRROR: also write bf16 copy to Cb.
// BOUNDED=false: caller guarantees M % 128 == 0 and N % 64 == 0 (branch-free
// staging + epilogue). Requires K % 32 == 0.
template <int ACT, bool MIRROR, bool BOUNDED>
__global__ __launch_bounds__(256)
void gemm_bf16_wmma(const unsigned short* __restrict__ A,
                    const unsigned short* __restrict__ W,
                    float* __restrict__ C, unsigned short* __restrict__ Cb,
                    const float* __restrict__ bias,
                    int M, int N, int K, int lda, int ldw, int ldc)
{
  constexpr int BM = 128, BN = 64, BK = 32;
  constexpr int LROW = BK + 8;                 // 40 halfs = 80B, 16B-aligned rows
  __shared__ alignas(16) unsigned short sA[BM * LROW];   // 10 KB
  __shared__ alignas(16) unsigned short sW[BN * LROW];   //  5 KB

  const int tid  = threadIdx.x;
  const int lane = tid & 31;
  const int wid  = tid >> 5;                   // 8 waves: 4 (M) x 2 (N)
  const int wm   = (wid >> 1) * 32;
  const int wn   = (wid & 1) * 32;
  const int lr   = lane & 15;
  const int half = lane >> 4;

  const int bm = blockIdx.y * BM;
  const int bn = blockIdx.x * BN;

  // Per-thread staging coordinates (pure 16B copies, no conversion).
  const int ar  = tid >> 1;                    // A: row, two chunks per thread
  const int ac  = (tid & 1) * 16;
  const bool aok = BOUNDED ? ((bm + ar) < M) : true;
  const int wr  = tid >> 2;                    // W: row, one chunk per thread
  const int wc  = (tid & 3) * 8;
  const bool wok = BOUNDED ? ((bn + wr) < N) : true;

  v8f acc[2][2] = {};

  for (int k0 = 0; k0 < K; k0 += BK) {
    const unsigned short* asrc = A + (size_t)(bm + ar) * lda + k0 + ac;
    unsigned short* adst = &sA[ar * LROW + ac];
    stage_chunk(asrc,     adst,     aok);
    stage_chunk(asrc + 8, adst + 8, aok);
    stage_chunk(W + (size_t)(bn + wr) * ldw + k0 + wc, &sW[wr * LROW + wc], wok);
#if ASYNC_COPY
    wait_async();
#endif
    __syncthreads();

    v16bf afrag[2], bfrag[2];
    #pragma unroll
    for (int mi = 0; mi < 2; ++mi)
      afrag[mi] = load_fragA(&sA[(wm + mi * 16 + lr) * LROW], half);
    #pragma unroll
    for (int ni = 0; ni < 2; ++ni)
      bfrag[ni] = load_fragB(&sW[(wn + ni * 16 + lr) * LROW], half);

    #pragma unroll
    for (int mi = 0; mi < 2; ++mi) {
      #pragma unroll
      for (int ni = 0; ni < 2; ++ni) {
        acc[mi][ni] = __builtin_amdgcn_wmma_f32_16x16x32_bf16(
            false, afrag[mi], false, bfrag[ni],
            (short)0, acc[mi][ni], false, false);
      }
    }
    __syncthreads();
  }

  // Epilogue. C/D layout: acc[r] holds row (r + 8*half), col = lane%16.
  #pragma unroll
  for (int mi = 0; mi < 2; ++mi) {
    #pragma unroll
    for (int ni = 0; ni < 2; ++ni) {
      int gcol = bn + wn + ni * 16 + lr;
      if (BOUNDED && gcol >= N) continue;
      float bv = (ACT == 1) ? bias[gcol] : 0.f;
      #pragma unroll
      for (int r = 0; r < 8; ++r) {
        int grow = bm + wm + mi * 16 + half * 8 + r;
        if (BOUNDED && grow >= M) continue;
        float v = acc[mi][ni][r];
        if (ACT == 1) {
          v += bv;
          v = (v > 20.f) ? v : log1pf(__expf(v));   // softplus
        }
        C[(size_t)grow * ldc + gcol] = v;
        if (MIRROR) Cb[(size_t)grow * ldc + gcol] = f2bf(v);
      }
    }
  }
}

// Elementwise fp32 -> bf16, 8 elements per thread (n % 2048 == 0 for all uses).
__global__ __launch_bounds__(256)
void to_bf16_kernel(const float* __restrict__ src, unsigned short* __restrict__ dst)
{
  const size_t i = (size_t)blockIdx.x * 256 + threadIdx.x;
  const float4* s = (const float4*)src + i * 2;
  float4 a = s[0], b = s[1];
  uint4 o;
  o.x = (unsigned)f2bf(a.x) | ((unsigned)f2bf(a.y) << 16);
  o.y = (unsigned)f2bf(a.z) | ((unsigned)f2bf(a.w) << 16);
  o.z = (unsigned)f2bf(b.x) | ((unsigned)f2bf(b.y) << 16);
  o.w = (unsigned)f2bf(b.z) | ((unsigned)f2bf(b.w) << 16);
  ((uint4*)dst)[i] = o;
}

// Depthwise causal conv1d (4 taps) + bias + SiLU -> bf16 u.
__global__ __launch_bounds__(256)
void conv_silu_kernel(const float* __restrict__ xz,
                      const float* __restrict__ conv_w,
                      const float* __restrict__ conv_b,
                      unsigned short* __restrict__ u_b)
{
  const size_t idx = (size_t)blockIdx.x * 256 + threadIdx.x;   // b*L*DI total
  const int d = (int)(idx % D_INNER_C);
  const size_t bl = idx / D_INNER_C;
  const int l = (int)(bl % SEQLEN_C);
  const size_t b = bl / SEQLEN_C;
  const size_t rowstride = 2 * (size_t)D_INNER_C;

  float s = conv_b[d];
  #pragma unroll
  for (int j = 0; j < D_CONV_C; ++j) {
    int ll = l - (D_CONV_C - 1) + j;
    if (ll >= 0)
      s += xz[(b * SEQLEN_C + ll) * rowstride + d] * conv_w[d * D_CONV_C + j];
  }
  float sig = 1.f / (1.f + __expf(-s));
  u_b[idx] = f2bf(s * sig);
}

// ---- Chunked selective scan (A diagonal => per-state affine composition) ----
// Pass 1: per (chunk, b, d): P[n] = prod_t dA_t[n]; S[n] = local scan, h_in=0.
__global__ __launch_bounds__(256)
void scan_chunk_reduce(const unsigned short* __restrict__ u_b,
                       const float* __restrict__ delta,
                       const float* __restrict__ xdbl,
                       const float* __restrict__ A_log,
                       float* __restrict__ P, float* __restrict__ S)
{
  const int gid = blockIdx.x * 256 + threadIdx.x;   // c*(B*DI) + b*DI + d
  const int d = gid % D_INNER_C;
  const int bc = gid / D_INNER_C;
  const int b = bc % BATCH_C;
  const int c = bc / BATCH_C;

  float Arow[D_STATE_C];
  #pragma unroll
  for (int n = 0; n < D_STATE_C; ++n) Arow[n] = -__expf(A_log[d * D_STATE_C + n]);
  float p[D_STATE_C], s_[D_STATE_C];
  #pragma unroll
  for (int n = 0; n < D_STATE_C; ++n) { p[n] = 1.f; s_[n] = 0.f; }

  const int t0 = c * CLEN_C;
  for (int t = t0; t < t0 + CLEN_C; ++t) {
    const size_t base = (size_t)b * SEQLEN_C + t;
    const float ut = bf2f(u_b[base * D_INNER_C + d]);
    const float dt = delta[base * D_INNER_C + d];
    const float* xd = xdbl + base * NDBL_C;
    const float du = dt * ut;
    #pragma unroll
    for (int n = 0; n < D_STATE_C; ++n) {
      float dA = __expf(dt * Arow[n]);
      p[n] *= dA;
      s_[n] = fmaf(dA, s_[n], du * xd[DT_RANK_C + n]);
    }
  }
  const size_t o = (size_t)gid * D_STATE_C;
  #pragma unroll
  for (int n = 0; n < D_STATE_C; ++n) { P[o + n] = p[n]; S[o + n] = s_[n]; }
}

// Pass 2: serial prefix over chunks; H[c] = state entering chunk c.
__global__ __launch_bounds__(256)
void scan_prefix(const float* __restrict__ P, const float* __restrict__ S,
                 float* __restrict__ H)
{
  const int gid = blockIdx.x * 256 + threadIdx.x;   // b*DI + d
  float h[D_STATE_C];
  #pragma unroll
  for (int n = 0; n < D_STATE_C; ++n) h[n] = 0.f;
  for (int c = 0; c < CHUNKS_C; ++c) {
    const size_t o = ((size_t)c * (BATCH_C * D_INNER_C) + gid) * D_STATE_C;
    #pragma unroll
    for (int n = 0; n < D_STATE_C; ++n) {
      H[o + n] = h[n];
      h[n] = fmaf(P[o + n], h[n], S[o + n]);
    }
  }
}

// Pass 3: replay chunk from true h_init; fused y = (scan + D*u)*silu(z) -> bf16.
__global__ __launch_bounds__(256)
void scan_chunk_apply(const unsigned short* __restrict__ u_b,
                      const float* __restrict__ delta,
                      const float* __restrict__ xdbl,
                      const float* __restrict__ xz,
                      const float* __restrict__ A_log,
                      const float* __restrict__ Dvec,
                      const float* __restrict__ H,
                      unsigned short* __restrict__ y_b)
{
  const int gid = blockIdx.x * 256 + threadIdx.x;   // c*(B*DI) + b*DI + d
  const int d = gid % D_INNER_C;
  const int bc = gid / D_INNER_C;
  const int b = bc % BATCH_C;
  const int c = bc / BATCH_C;

  float Arow[D_STATE_C], h[D_STATE_C];
  #pragma unroll
  for (int n = 0; n < D_STATE_C; ++n) Arow[n] = -__expf(A_log[d * D_STATE_C + n]);
  const size_t ho = (size_t)gid * D_STATE_C;
  #pragma unroll
  for (int n = 0; n < D_STATE_C; ++n) h[n] = H[ho + n];

  const float Dd = Dvec[d];
  const size_t rowXZ = 2 * (size_t)D_INNER_C;
  const int t0 = c * CLEN_C;
  for (int t = t0; t < t0 + CLEN_C; ++t) {
    const size_t base = (size_t)b * SEQLEN_C + t;
    const float ut = bf2f(u_b[base * D_INNER_C + d]);
    const float dt = delta[base * D_INNER_C + d];
    const float* xd = xdbl + base * NDBL_C;
    const float du = dt * ut;
    float yt = 0.f;
    #pragma unroll
    for (int n = 0; n < D_STATE_C; ++n) {
      float dA = __expf(dt * Arow[n]);
      h[n] = fmaf(dA, h[n], du * xd[DT_RANK_C + n]);              // B_t
      yt = fmaf(h[n], xd[DT_RANK_C + D_STATE_C + n], yt);         // C_t
    }
    const float zt = xz[base * rowXZ + D_INNER_C + d];
    const float g = zt / (1.f + __expf(-zt));                     // silu(z)
    y_b[base * D_INNER_C + d] = f2bf((yt + Dd * ut) * g);
  }
}

extern "C" void kernel_launch(void* const* d_in, const int* in_sizes, int n_in,
                              void* d_out, int out_size, void* d_ws, size_t ws_size,
                              hipStream_t stream) {
  (void)in_sizes; (void)n_in; (void)out_size; (void)ws_size;

  const float* x      = (const float*)d_in[0];
  const float* W_in   = (const float*)d_in[1];
  const float* conv_w = (const float*)d_in[2];
  const float* conv_b = (const float*)d_in[3];
  const float* W_x    = (const float*)d_in[4];
  const float* W_dt   = (const float*)d_in[5];
  const float* b_dt   = (const float*)d_in[6];
  const float* A_log  = (const float*)d_in[7];
  const float* Dvec   = (const float*)d_in[8];
  const float* W_out  = (const float*)d_in[9];
  float* out = (float*)d_out;

  const size_t nTok   = (size_t)BATCH_C * SEQLEN_C;           // 4096
  const size_t nXZ    = nTok * 2 * D_INNER_C;                 // 33.6M
  const size_t nU     = nTok * D_INNER_C;                     // 16.8M
  const size_t nXDBL  = nTok * NDBL_C;                        // 0.66M
  const size_t nPS    = (size_t)CHUNKS_C * BATCH_C * D_INNER_C * D_STATE_C; // 2.1M

  // fp32 region
  float* fws = (float*)d_ws;
  size_t o = 0;
  float* xz    = fws + o; o += nXZ;
  float* xdbl  = fws + o; o += nXDBL;
  float* delta = fws + o; o += nU;
  float* P     = fws + o; o += nPS;
  float* S     = fws + o; o += nPS;
  float* H     = fws + o; o += nPS;
  // bf16 region
  unsigned short* hws = (unsigned short*)(fws + o);
  size_t ho = 0;
  unsigned short* xb     = hws + ho; ho += nTok * D_MODEL_C;
  unsigned short* W_inb  = hws + ho; ho += (size_t)2 * D_INNER_C * D_MODEL_C;
  unsigned short* W_xb   = hws + ho; ho += (size_t)NDBL_C * D_INNER_C;
  unsigned short* W_dtb  = hws + ho; ho += (size_t)D_INNER_C * DT_RANK_C;
  unsigned short* W_outb = hws + ho; ho += (size_t)D_MODEL_C * D_INNER_C;
  unsigned short* u_b    = hws + ho; ho += nU;
  unsigned short* xdbl_b = hws + ho; ho += nXDBL;
  unsigned short* y_b    = hws + ho; ho += nU;

  const int M = (int)nTok;
  const dim3 blk(256);

  // 0) operand conversions (each size % 2048 == 0)
  to_bf16_kernel<<<(nTok * D_MODEL_C) / 2048, blk, 0, stream>>>(x, xb);
  to_bf16_kernel<<<((size_t)2 * D_INNER_C * D_MODEL_C) / 2048, blk, 0, stream>>>(W_in, W_inb);
  to_bf16_kernel<<<((size_t)NDBL_C * D_INNER_C) / 2048, blk, 0, stream>>>(W_x, W_xb);
  to_bf16_kernel<<<((size_t)D_INNER_C * DT_RANK_C) / 2048, blk, 0, stream>>>(W_dt, W_dtb);
  to_bf16_kernel<<<((size_t)D_MODEL_C * D_INNER_C) / 2048, blk, 0, stream>>>(W_out, W_outb);

  // 1) xz = x @ W_in^T : M=4096, N=8192, K=2048 (aligned -> unguarded)
  gemm_bf16_wmma<0, false, false><<<dim3((2 * D_INNER_C + 63) / 64, (M + 127) / 128), blk, 0, stream>>>(
      xb, W_inb, xz, nullptr, nullptr, M, 2 * D_INNER_C, D_MODEL_C,
      D_MODEL_C, D_MODEL_C, 2 * D_INNER_C);

  // 2) u = silu(conv(x_) + conv_b) -> bf16
  conv_silu_kernel<<<(BATCH_C * SEQLEN_C * D_INNER_C) / 256, blk, 0, stream>>>(
      xz, conv_w, conv_b, u_b);

  // 3) x_dbl = u @ W_x^T : M=4096, N=160, K=4096 (edge N -> guarded; bf16 mirror)
  gemm_bf16_wmma<0, true, true><<<dim3((NDBL_C + 63) / 64, (M + 127) / 128), blk, 0, stream>>>(
      u_b, W_xb, xdbl, xdbl_b, nullptr, M, NDBL_C, D_INNER_C,
      D_INNER_C, D_INNER_C, NDBL_C);

  // 4) delta = softplus(x_dbl[:, :128] @ W_dt^T + b_dt) : M=4096, N=4096, K=128
  gemm_bf16_wmma<1, false, false><<<dim3((D_INNER_C + 63) / 64, (M + 127) / 128), blk, 0, stream>>>(
      xdbl_b, W_dtb, delta, nullptr, b_dt, M, D_INNER_C, DT_RANK_C,
      NDBL_C, DT_RANK_C, D_INNER_C);

  // 5) chunked selective scan
  scan_chunk_reduce<<<(CHUNKS_C * BATCH_C * D_INNER_C) / 256, blk, 0, stream>>>(
      u_b, delta, xdbl, A_log, P, S);
  scan_prefix<<<(BATCH_C * D_INNER_C) / 256, blk, 0, stream>>>(P, S, H);
  scan_chunk_apply<<<(CHUNKS_C * BATCH_C * D_INNER_C) / 256, blk, 0, stream>>>(
      u_b, delta, xdbl, xz, A_log, Dvec, H, y_b);

  // 6) out = y @ W_out^T : M=4096, N=2048, K=4096 (aligned -> unguarded)
  gemm_bf16_wmma<0, false, false><<<dim3((D_MODEL_C + 63) / 64, (M + 127) / 128), blk, 0, stream>>>(
      y_b, W_outb, out, nullptr, nullptr, M, D_MODEL_C, D_INNER_C,
      D_INNER_C, D_INNER_C, D_MODEL_C);
}
